// HasteGRU_40398462386411
// MI455X (gfx1250) — matrix operand order repaired and checked by reading, not compiled, not run on hardware
//
#include <hip/hip_runtime.h>
#include <hip/hip_bf16.h>
#include <stdint.h>

typedef __bf16 bfx;
typedef __attribute__((ext_vector_type(16))) __bf16 v16bf;
typedef __attribute__((ext_vector_type(8)))  __bf16 v8bf;
typedef __attribute__((ext_vector_type(8)))  float  v8f;

__device__ __forceinline__ unsigned short f2bf(float f) {
    unsigned u = __float_as_uint(f);
    u += 0x7fffu + ((u >> 16) & 1u);        // round-to-nearest-even
    return (unsigned short)(u >> 16);
}

// -------------------- conversion / init kernels --------------------
__global__ __launch_bounds__(256) void cvt_f32_to_bf16(const float* __restrict__ src,
                                                       unsigned short* __restrict__ dst, int n) {
    int i = blockIdx.x * 256 + threadIdx.x;
    if (i < n) dst[i] = f2bf(src[i]);
}

// src: [K,N] row-major f32  ->  dst: [N,K] bf16 (column-major of src)
__global__ __launch_bounds__(256) void transpose_cvt(const float* __restrict__ src,
                                                     unsigned short* __restrict__ dst,
                                                     int K, int N) {
    int i = blockIdx.x * 256 + threadIdx.x;
    if (i >= K * N) return;
    int k = i / N, n = i % N;
    dst[(size_t)n * K + k] = f2bf(src[i]);
}

__global__ __launch_bounds__(256) void init_h(float* __restrict__ h,
                                              unsigned short* __restrict__ hb, int n) {
    int i = blockIdx.x * 256 + threadIdx.x;
    if (i < n) { h[i] = 0.f; hb[i] = 0; }
}

// -------------------- WMMA bf16 GEMM: C[M,N] = A[M,K] * Bt[N,K]^T --------------------
// A row-major bf16, Bt is B pre-transposed ([N,K] bf16), C f32 row-major (overwritten).
// One wave32 computes a 32x64 tile: 2 A-fragments x 4 B-fragments -> 8
// v_wmma_f32_16x16x32_bf16 per K=32 step, 12 b128 loads (1.5 loads/WMMA).
__global__ __launch_bounds__(256) void gemm_bf16_wmma(const bfx* __restrict__ A,
                                                      const bfx* __restrict__ Bt,
                                                      float* __restrict__ C,
                                                      int M, int N, int K) {
    const int lane = threadIdx.x & 31;
    const int wave = threadIdx.x >> 5;
    const int nT   = N >> 6;                 // # of 64-wide N tiles
    const int total = (M >> 5) * nT;         // 32-row M tiles
    const int tile = blockIdx.x * 8 + wave;  // wave-uniform
    if (tile >= total) return;               // whole-wave exit: EXEC stays all-1s

    const int mBase = (tile / nT) << 5;
    const int nBase = (tile % nT) << 6;
    const int half  = lane >> 4;             // 0: lanes 0-15, 1: lanes 16-31
    const int l16   = lane & 15;

    // A fragments: rows mBase+l16 and mBase+16+l16; contiguous K run 16*half..+15
    const bfx* aP0 = A  + (size_t)(mBase +  0 + l16) * K + (half << 4);
    const bfx* aP1 = A  + (size_t)(mBase + 16 + l16) * K + (half << 4);
    // B fragments: column n = nBase+16j+l16 of B == row of Bt; contiguous K run 16*half..+15
    const bfx* bP0 = Bt + (size_t)(nBase +  0 + l16) * K + (half << 4);
    const bfx* bP1 = Bt + (size_t)(nBase + 16 + l16) * K + (half << 4);
    const bfx* bP2 = Bt + (size_t)(nBase + 32 + l16) * K + (half << 4);
    const bfx* bP3 = Bt + (size_t)(nBase + 48 + l16) * K + (half << 4);

    v8f acc[2][4] = {};
    union AF { v16bf v; v8bf h[2]; };

    for (int k = 0; k < K; k += 32) {
        AF a0, a1;
        a0.h[0] = *(const v8bf*)(aP0 + k);
        a0.h[1] = *(const v8bf*)(aP0 + k + 8);
        a1.h[0] = *(const v8bf*)(aP1 + k);
        a1.h[1] = *(const v8bf*)(aP1 + k + 8);
        v16bf b0 = *(const v16bf*)(bP0 + k);
        v16bf b1 = *(const v16bf*)(bP1 + k);
        v16bf b2 = *(const v16bf*)(bP2 + k);
        v16bf b3 = *(const v16bf*)(bP3 + k);
        acc[0][0] = __builtin_amdgcn_wmma_f32_16x16x32_bf16(false, a0.v, false, b0, (short)0, acc[0][0], false, false);
        acc[0][1] = __builtin_amdgcn_wmma_f32_16x16x32_bf16(false, a0.v, false, b1, (short)0, acc[0][1], false, false);
        acc[0][2] = __builtin_amdgcn_wmma_f32_16x16x32_bf16(false, a0.v, false, b2, (short)0, acc[0][2], false, false);
        acc[0][3] = __builtin_amdgcn_wmma_f32_16x16x32_bf16(false, a0.v, false, b3, (short)0, acc[0][3], false, false);
        acc[1][0] = __builtin_amdgcn_wmma_f32_16x16x32_bf16(false, a1.v, false, b0, (short)0, acc[1][0], false, false);
        acc[1][1] = __builtin_amdgcn_wmma_f32_16x16x32_bf16(false, a1.v, false, b1, (short)0, acc[1][1], false, false);
        acc[1][2] = __builtin_amdgcn_wmma_f32_16x16x32_bf16(false, a1.v, false, b2, (short)0, acc[1][2], false, false);
        acc[1][3] = __builtin_amdgcn_wmma_f32_16x16x32_bf16(false, a1.v, false, b3, (short)0, acc[1][3], false, false);
    }

    // C/D layout: VGPR r -> row (tile row base) + 8*half + r, col nBase + l16 (+16j per acc)
    #pragma unroll
    for (int i = 0; i < 2; ++i) {
        #pragma unroll
        for (int r = 0; r < 8; ++r) {
            float* cr = C + (size_t)(mBase + (i << 4) + (half << 3) + r) * N + nBase + l16;
            cr[0]  = acc[i][0][r];
            cr[16] = acc[i][1][r];
            cr[32] = acc[i][2][r];
            cr[48] = acc[i][3][r];
        }
    }
}

// -------------------- row LayerNorm (in place): x = LN(x)*gamma + beta --------------------
__global__ __launch_bounds__(256) void layernorm_rows(float* __restrict__ X,
                                                      const float* __restrict__ gamma,
                                                      const float* __restrict__ beta,
                                                      int C) {
    __shared__ float rs[256], rss[256];
    float* x = X + (size_t)blockIdx.x * C;
    float s = 0.f, ss = 0.f;
    for (int i = threadIdx.x; i < C; i += 256) { float v = x[i]; s += v; ss += v * v; }
    rs[threadIdx.x] = s; rss[threadIdx.x] = ss;
    __syncthreads();
    for (int o = 128; o > 0; o >>= 1) {
        if (threadIdx.x < o) {
            rs[threadIdx.x]  += rs[threadIdx.x + o];
            rss[threadIdx.x] += rss[threadIdx.x + o];
        }
        __syncthreads();
    }
    float mean = rs[0] / C;
    float var  = rss[0] / C - mean * mean;
    float inv  = rsqrtf(var + 1e-5f);
    for (int i = threadIdx.x; i < C; i += 256)
        x[i] = (x[i] - mean) * inv * gamma[i] + beta[i];
}

// -------------------- fused GRU gates; also emits bf16 h (and optional y_t bf16) --------------------
__global__ __launch_bounds__(256) void gru_gates(const float* __restrict__ Wx,  // [B*T,3H] LN'ed + b
                                                 int t, int T,
                                                 const float* __restrict__ Rh,  // [B,3H]  LN'ed + rb
                                                 float* __restrict__ h,         // [B,H] in/out
                                                 unsigned short* __restrict__ hb,  // [B,H] bf16 out
                                                 unsigned short* __restrict__ yb,  // [B*T,H] bf16 out or null
                                                 int B, int H) {
    int idx = blockIdx.x * 256 + threadIdx.x;
    if (idx >= B * H) return;
    int b = idx / H, i = idx - b * H;
    const float* wx = Wx + (size_t)(b * T + t) * (3 * H);
    const float* rh = Rh + (size_t)b * (3 * H);
    float zx = wx[i], rx = wx[H + i], gx = wx[2 * H + i];
    float zh = rh[i], rr = rh[H + i], gh = rh[2 * H + i];
    float z = 1.f / (1.f + __expf(-(zx + zh)));
    float r = 1.f / (1.f + __expf(-(rx + rr)));
    float g = tanhf(gx + r * gh);
    float hn = z * h[idx] + (1.f - z) * g;
    h[idx] = hn;
    unsigned short h16 = f2bf(hn);
    hb[idx] = h16;
    if (yb) yb[(size_t)(b * T + t) * H + i] = h16;
}

// -------------------- head: out = fcacc + fc_b ; append final hidden --------------------
__global__ __launch_bounds__(256) void finalize(const float* __restrict__ fcacc,
                                                const float* __restrict__ fcb,
                                                const float* __restrict__ h,
                                                float* __restrict__ out,
                                                int O, int BO, int BH) {
    int i = blockIdx.x * 256 + threadIdx.x;
    if (i < BO) out[i] = fcacc[i] + fcb[i % O];
    if (i < BH) out[BO + i] = h[i];
}

// -------------------- host orchestration --------------------
extern "C" void kernel_launch(void* const* d_in, const int* in_sizes, int n_in,
                              void* d_out, int out_size, void* d_ws, size_t ws_size,
                              hipStream_t stream) {
    const float* x   = (const float*)d_in[0];
    const float* k0  = (const float*)d_in[1];
    const float* rk0 = (const float*)d_in[2];
    const float* b0  = (const float*)d_in[3];
    const float* rb0 = (const float*)d_in[4];
    const float* gx0 = (const float*)d_in[5];
    const float* gh0 = (const float*)d_in[6];
    const float* k1  = (const float*)d_in[7];
    const float* rk1 = (const float*)d_in[8];
    const float* b1  = (const float*)d_in[9];
    const float* rb1 = (const float*)d_in[10];
    const float* gx1 = (const float*)d_in[11];
    const float* gh1 = (const float*)d_in[12];
    const float* fcw = (const float*)d_in[13];
    const float* fcb = (const float*)d_in[14];

    const int B = 64, T = 128, D = 512, H = 1024, O = 512;
    const int H3 = 3 * H;

    char* ws = (char*)d_ws;
    size_t off = 0;
    auto alloc = [&](size_t bytes) -> void* {
        void* p = ws + off;
        off = (off + bytes + 255) & ~(size_t)255;
        return p;
    };

    unsigned short* xb    = (unsigned short*)alloc((size_t)B * T * D * 2);   // 8 MB
    unsigned short* k0t   = (unsigned short*)alloc((size_t)H3 * D * 2);      // 3 MB
    unsigned short* rk0t  = (unsigned short*)alloc((size_t)H3 * H * 2);      // 6 MB
    unsigned short* k1t   = (unsigned short*)alloc((size_t)H3 * H * 2);      // 6 MB
    unsigned short* rk1t  = (unsigned short*)alloc((size_t)H3 * H * 2);      // 6 MB
    unsigned short* fcwt  = (unsigned short*)alloc((size_t)O * H * 2);       // 1 MB
    float*          Wx    = (float*)alloc((size_t)B * T * H3 * 4);           // 96 MB, reused both layers
    float*          Rh    = (float*)alloc((size_t)B * H3 * 4);               // 0.75 MB
    float*          hbuf  = (float*)alloc((size_t)B * H * 4);                // 0.25 MB
    unsigned short* hb    = (unsigned short*)alloc((size_t)B * H * 2);
    unsigned short* y0b   = (unsigned short*)alloc((size_t)B * T * H * 2);   // 16 MB
    float*          fcacc = (float*)alloc((size_t)B * O * 4);

    auto gemm = [&](const unsigned short* A, const unsigned short* Bt, float* C,
                    int M, int N, int K) {
        int tiles  = (M >> 5) * (N >> 6);
        int blocks = (tiles + 7) / 8;
        gemm_bf16_wmma<<<blocks, 256, 0, stream>>>((const bfx*)A, (const bfx*)Bt, C, M, N, K);
    };

    { int n = B * T * D; cvt_f32_to_bf16<<<(n + 255) / 256, 256, 0, stream>>>(x, xb, n); }
    { int n = D * H3; transpose_cvt<<<(n + 255) / 256, 256, 0, stream>>>(k0,  k0t,  D, H3); }
    { int n = H * H3; transpose_cvt<<<(n + 255) / 256, 256, 0, stream>>>(rk0, rk0t, H, H3); }
    { int n = H * H3; transpose_cvt<<<(n + 255) / 256, 256, 0, stream>>>(k1,  k1t,  H, H3); }
    { int n = H * H3; transpose_cvt<<<(n + 255) / 256, 256, 0, stream>>>(rk1, rk1t, H, H3); }
    { int n = H * O;  transpose_cvt<<<(n + 255) / 256, 256, 0, stream>>>(fcw, fcwt, H, O); }
    { int n = B * H;  init_h<<<(n + 255) / 256, 256, 0, stream>>>(hbuf, hb, n); }

    // ---- layer 0 ----
    gemm(xb, k0t, Wx, B * T, H3, D);
    layernorm_rows<<<B * T, 256, 0, stream>>>(Wx, gx0, b0, H3);
    for (int t = 0; t < T; ++t) {
        gemm(hb, rk0t, Rh, B, H3, H);
        layernorm_rows<<<B, 256, 0, stream>>>(Rh, gh0, rb0, H3);
        gru_gates<<<(B * H + 255) / 256, 256, 0, stream>>>(Wx, t, T, Rh, hbuf, hb, y0b, B, H);
    }
    // ---- layer 1 (h threaded across layers, Wx buffer reused) ----
    gemm(y0b, k1t, Wx, B * T, H3, H);
    layernorm_rows<<<B * T, 256, 0, stream>>>(Wx, gx1, b1, H3);
    for (int t = 0; t < T; ++t) {
        gemm(hb, rk1t, Rh, B, H3, H);
        layernorm_rows<<<B, 256, 0, stream>>>(Rh, gh1, rb1, H3);
        gru_gates<<<(B * H + 255) / 256, 256, 0, stream>>>(Wx, t, T, Rh, hbuf, hb,
                                                           (unsigned short*)nullptr, B, H);
    }
    // ---- head: y[:, -1, :] == final h of layer 1 ----
    gemm(hb, fcwt, fcacc, B, O, H);
    { int n = B * H;  // covers both B*O (32768) and B*H (65536)
      finalize<<<(n + 255) / 256, 256, 0, stream>>>(fcacc, fcb, hbuf, (float*)d_out,
                                                    O, B * O, B * H); }
}
